// EulerAttention_75892072121024
// MI455X (gfx1250) — matrix operand order repaired and checked by reading, not compiled, not run on hardware
//
#include <hip/hip_runtime.h>
#include <hip/hip_bf16.h>
#include <math.h>

typedef __bf16 bf16_t;
typedef __bf16 v16bf __attribute__((ext_vector_type(16)));
typedef __bf16 v8bf  __attribute__((ext_vector_type(8)));
typedef float  v8f   __attribute__((ext_vector_type(8)));
typedef unsigned int u32x4 __attribute__((ext_vector_type(4)));
typedef int          i32x4 __attribute__((ext_vector_type(4)));
typedef int          i32x8 __attribute__((ext_vector_type(8)));

constexpr int BATCH = 2;
constexpr int S     = 4096;
constexpr int D     = 1024;
constexpr int NROW  = BATCH * S;   // 8192
constexpr int CH    = 512;         // query-row chunk for score materialization

// ---------------- WMMA fragment loaders (gfx1250 wave32 layouts) ----------------

// A-matrix 16x32 bf16: lanes 0-15 -> row m0+lane, K = kb+{0..7} and kb+{16..23}
//                      lanes 16-31 -> row m0+lane-16, K = kb+{8..15} and kb+{24..31}
__device__ __forceinline__ v16bf load_a_bf16(const bf16_t* base, int ld, int m0, int kb, int lane) {
    int row  = m0 + (lane & 15);
    int koff = kb + ((lane >> 4) << 3);
    const bf16_t* p = base + (size_t)row * ld + koff;
    union { v16bf v; v8bf h[2]; } u;
    u.h[0] = *(const v8bf*)(p);
    u.h[1] = *(const v8bf*)(p + 16);
    return u.v;
}

// Same A layout, converting f32 source on the fly
__device__ __forceinline__ v16bf load_a_f32(const float* base, int ld, int m0, int kb, int lane) {
    int row  = m0 + (lane & 15);
    int koff = kb + ((lane >> 4) << 3);
    const float* p = base + (size_t)row * ld + koff;
    v16bf r;
#pragma unroll
    for (int i = 0; i < 8; ++i) r[i] = (bf16_t)p[i];
#pragma unroll
    for (int i = 0; i < 8; ++i) r[8 + i] = (bf16_t)p[16 + i];
    return r;
}

// B-matrix 32x16 bf16, `base` stores B^T row-major (row c = column c of B):
// lanes 0-15 -> col n0+lane, K = kb+{0..7} and kb+{16..23} ; lanes 16-31 -> col n0+lane-16, K = kb+{8..15} and kb+{24..31}
__device__ __forceinline__ v16bf load_b_bf16(const bf16_t* base, int ld, int n0, int kb, int lane) {
    int col  = n0 + (lane & 15);
    int koff = kb + ((lane >> 4) << 3);
    const bf16_t* p = base + (size_t)col * ld + koff;
    union { v16bf v; v8bf h[2]; } u;
    u.h[0] = *(const v8bf*)(p);
    u.h[1] = *(const v8bf*)(p + 16);
    return u.v;
}

__device__ __forceinline__ v8f wmma_bf16(v16bf a, v16bf b, v8f c) {
    return __builtin_amdgcn_wmma_f32_16x16x32_bf16(false, a, false, b, (short)0, c, false, false);
}

// ---------------- Tensor Data Mover: 2D panel -> LDS ----------------
// D# per CDNA5 ISA 8.3/8.4: group0 {count=1, lds_addr, global_addr, type=2},
// group1 {data_size=2B, tensor_dim0=D, tensor_dim1=big, tile_dim0=32(K), tile_dim1=rows,
//         tensor_dim0_stride=ld}. Groups 2/3 zero (2D tensor).
__device__ __forceinline__ void tdm_load_panel(const bf16_t* gptr, unsigned lds_byte_off,
                                               int rows, int ld_elems) {
    unsigned long long ga = (unsigned long long)(uintptr_t)gptr;
    u32x4 g0;
    g0[0] = 1u;                                             // count=1, user mode
    g0[1] = lds_byte_off;                                   // lds_addr
    g0[2] = (unsigned)(ga & 0xFFFFFFFFu);                   // global_addr[31:0]
    g0[3] = (unsigned)((ga >> 32) & 0x01FFFFFFu) | (2u << 30); // addr[56:32] | type=2
    i32x8 g1;
    g1[0] = (1 << 16);                                      // data_size = 1 -> 2 bytes
    g1[1] = (int)(((unsigned)D & 0xFFFFu) << 16);           // tensor_dim0[15:0] @bits63:48
    g1[2] = (int)(((unsigned)(D >> 16)) | (0xFFFFu << 16)); // dim0 hi | tensor_dim1 lo = 65535
    g1[3] = (int)(0u | (32u << 16));                        // dim1 hi=0 | tile_dim0 = 32
    g1[4] = rows & 0xFFFF;                                  // tile_dim1 = rows, tile_dim2 = 0
    g1[5] = ld_elems;                                       // tensor_dim0_stride[31:0]
    g1[6] = 0;
    g1[7] = 0;
    i32x4 z4 = {0, 0, 0, 0};
#if defined(__clang_major__) && (__clang_major__ >= 23)
    i32x8 z8 = {0, 0, 0, 0, 0, 0, 0, 0};
    __builtin_amdgcn_tensor_load_to_lds(g0, g1, z4, z4, z8, 0);
#else
    __builtin_amdgcn_tensor_load_to_lds(g0, g1, z4, z4, 0);
#endif
}

// ---------------- kernels ----------------

__global__ void cvt_f32_to_bf16(const float* __restrict__ src, bf16_t* __restrict__ dst, int n) {
    int i = blockIdx.x * blockDim.x + threadIdx.x;
    if (i < n) dst[i] = (bf16_t)src[i];
}

// Fused QKV projection + theta + cos/sin. Wave computes 16x32 (2 N-tiles) for Q,K,V:
// one A fragment feeds 6 WMMAs. M = NROW, N = D, K = D.
__global__ void qkv_theta_kernel(const float* __restrict__ x,
                                 const bf16_t* __restrict__ Wqb,
                                 const bf16_t* __restrict__ Wkb,
                                 const bf16_t* __restrict__ Wvb,
                                 const float* __restrict__ bq,
                                 const float* __restrict__ bk,
                                 const float* __restrict__ bv,
                                 const float* __restrict__ phase_bias,
                                 bf16_t* __restrict__ cq, bf16_t* __restrict__ sq,
                                 bf16_t* __restrict__ ck, bf16_t* __restrict__ sk,
                                 bf16_t* __restrict__ vbT) {
    int w    = (blockIdx.x * blockDim.x + threadIdx.x) >> 5;
    int lane = threadIdx.x & 31;
    int n0 = (w & 31) << 5;           // 32 N-super-tiles of width 32
    int m0 = (w >> 5) << 4;           // 512 M-tiles

    v8f accQ[2] = {}, accK[2] = {}, accV[2] = {};
    for (int kb = 0; kb < D; kb += 32) {
        v16bf a = load_a_f32(x, D, m0, kb, lane);
#pragma unroll
        for (int h = 0; h < 2; ++h) {
            v16bf bQ = load_b_bf16(Wqb, D, n0 + 16 * h, kb, lane);
            v16bf bK = load_b_bf16(Wkb, D, n0 + 16 * h, kb, lane);
            v16bf bV = load_b_bf16(Wvb, D, n0 + 16 * h, kb, lane);
            accQ[h] = wmma_bf16(a, bQ, accQ[h]);
            accK[h] = wmma_bf16(a, bK, accK[h]);
            accV[h] = wmma_bf16(a, bV, accV[h]);
        }
    }

    int roff = (lane >> 4) << 3;
#pragma unroll
    for (int h = 0; h < 2; ++h) {
        int col = n0 + 16 * h + (lane & 15);
        float iwl = 1.0f / ((float)(col + 1) * (6.2831853071795864769f / (float)D) + 1e-8f);
        float pb  = phase_bias[col];
        float bQs = bq[col], bKs = bk[col], bVs = bv[col];
#pragma unroll
        for (int r = 0; r < 8; ++r) {
            int row = m0 + r + roff;                 // global row over B*S
            int b = row >> 12;
            int t = row & (S - 1);
            size_t idx = (size_t)row * D + col;
            float tq = fmaf(accQ[h][r] + bQs, iwl, pb);
            float tk = fmaf(accK[h][r] + bKs, iwl, pb);
            float sv, cv;
            __sincosf(tq, &sv, &cv);
            cq[idx] = (bf16_t)cv;  sq[idx] = (bf16_t)sv;
            __sincosf(tk, &sv, &cv);
            ck[idx] = (bf16_t)cv;  sk[idx] = (bf16_t)sv;
            vbT[((size_t)b * D + col) * S + t] = (bf16_t)(accV[h][r] + bVs);
        }
    }
}

// Scores: sim[CH x S] = (cq@ck^T + sq@sk^T)/sqrt(D).
// Block tile 64x128, 8 waves (2x4), each wave 32x32 (2Mt x 2Nt).
// K staged in LDS by the Tensor Data Mover, double-buffered, KC=32 per chunk.
constexpr int SC_BM = 64, SC_BN = 128, SC_KC = 32;
constexpr int L_AC = 0;                         // cq panel  64x32
constexpr int L_AS = L_AC + SC_BM * SC_KC;      // sq panel  64x32
constexpr int L_BC = L_AS + SC_BM * SC_KC;      // ck panel 128x32
constexpr int L_BS = L_BC + SC_BN * SC_KC;      // sk panel 128x32
constexpr int L_BUF = L_BS + SC_BN * SC_KC;     // 12288 elems = 24KB per buffer

__global__ void scores_kernel(const bf16_t* __restrict__ cq, const bf16_t* __restrict__ sq,
                              const bf16_t* __restrict__ ck, const bf16_t* __restrict__ sk,
                              float* __restrict__ sim, int b, int m_start) {
    __shared__ __align__(128) bf16_t smem[2 * L_BUF];   // 48KB, double buffered

    int lane = threadIdx.x & 31;
    int wv   = threadIdx.x >> 5;      // 0..7
    int wm   = wv >> 2;               // 0..1 -> 32-row strip
    int wn   = wv & 3;                // 0..3 -> 32-col strip

    int nb_n    = S / SC_BN;                              // 32
    int block_m = (blockIdx.x / nb_n) * SC_BM;            // chunk-local
    int block_n = (blockIdx.x % nb_n) * SC_BN;

    size_t boff = (size_t)b * S * D;
    const bf16_t* cqb = cq + boff + (size_t)(m_start + block_m) * D;
    const bf16_t* sqb = sq + boff + (size_t)(m_start + block_m) * D;
    const bf16_t* ckb = ck + boff + (size_t)block_n * D;
    const bf16_t* skb = sk + boff + (size_t)block_n * D;

    v8f acc[2][2] = {};
    constexpr int NCH = D / SC_KC;    // 32 K-chunks

    if (wv == 0) {   // wave 0 drives the TDM pipeline (EXEC-independent DMA)
        tdm_load_panel(cqb, (unsigned)(L_AC * 2), SC_BM, D);
        tdm_load_panel(sqb, (unsigned)(L_AS * 2), SC_BM, D);
        tdm_load_panel(ckb, (unsigned)(L_BC * 2), SC_BN, D);
        tdm_load_panel(skb, (unsigned)(L_BS * 2), SC_BN, D);
    }

    int bsel = 0;
    for (int kc = 0; kc < NCH; ++kc) {
        if (wv == 0) {
            if (kc + 1 < NCH) {
                unsigned nb = (unsigned)((bsel ^ 1) * L_BUF * 2);
                int kb = (kc + 1) * SC_KC;
                tdm_load_panel(cqb + kb, nb + L_AC * 2, SC_BM, D);
                tdm_load_panel(sqb + kb, nb + L_AS * 2, SC_BM, D);
                tdm_load_panel(ckb + kb, nb + L_BC * 2, SC_BN, D);
                tdm_load_panel(skb + kb, nb + L_BS * 2, SC_BN, D);
                __builtin_amdgcn_s_wait_tensorcnt(4);  // current buffer's 4 loads done
            } else {
                __builtin_amdgcn_s_wait_tensorcnt(0);
            }
        }
        __syncthreads();               // buffer bsel ready for all waves

        const bf16_t* base = smem + bsel * L_BUF;
#pragma unroll
        for (int mt = 0; mt < 2; ++mt) {
            v16bf ac = load_a_bf16(base + L_AC, SC_KC, wm * 32 + mt * 16, 0, lane);
            v16bf as = load_a_bf16(base + L_AS, SC_KC, wm * 32 + mt * 16, 0, lane);
#pragma unroll
            for (int nt = 0; nt < 2; ++nt) {
                v16bf bc = load_b_bf16(base + L_BC, SC_KC, wn * 32 + nt * 16, 0, lane);
                v16bf bs = load_b_bf16(base + L_BS, SC_KC, wn * 32 + nt * 16, 0, lane);
                acc[mt][nt] = wmma_bf16(ac, bc, acc[mt][nt]);
                acc[mt][nt] = wmma_bf16(as, bs, acc[mt][nt]);
            }
        }
        __syncthreads();               // all reads done before buffer reuse
        bsel ^= 1;
    }

    int roff = (lane >> 4) << 3;
#pragma unroll
    for (int mt = 0; mt < 2; ++mt) {
#pragma unroll
        for (int nt = 0; nt < 2; ++nt) {
            int col = block_n + wn * 32 + nt * 16 + (lane & 15);
#pragma unroll
            for (int r = 0; r < 8; ++r) {
                int lr = block_m + wm * 32 + mt * 16 + r + roff;
                sim[(size_t)lr * S + col] = acc[mt][nt][r] * 0.03125f;  // 1/sqrt(1024)
            }
        }
    }
}

// Row softmax over S=4096, writing bf16 probabilities. One 256-thread block per row.
__global__ void softmax_kernel(const float* __restrict__ sim, bf16_t* __restrict__ P) {
    int row = blockIdx.x;
    int t   = threadIdx.x;
    const float* sr = sim + (size_t)row * S;

    float v[16];
#pragma unroll
    for (int i = 0; i < 16; ++i) v[i] = sr[t + 256 * i];

    float m = -3.4e38f;
#pragma unroll
    for (int i = 0; i < 16; ++i) m = fmaxf(m, v[i]);

    __shared__ float red[256];
    red[t] = m; __syncthreads();
    for (int s = 128; s > 0; s >>= 1) {
        if (t < s) red[t] = fmaxf(red[t], red[t + s]);
        __syncthreads();
    }
    m = red[0]; __syncthreads();

    float sum = 0.0f;
#pragma unroll
    for (int i = 0; i < 16; ++i) { v[i] = __expf(v[i] - m); sum += v[i]; }
    red[t] = sum; __syncthreads();
    for (int s = 128; s > 0; s >>= 1) {
        if (t < s) red[t] += red[t + s];
        __syncthreads();
    }
    float inv = 1.0f / red[0];

    bf16_t* pr = P + (size_t)row * S;
#pragma unroll
    for (int i = 0; i < 16; ++i) pr[t + 256 * i] = (bf16_t)(v[i] * inv);
}

// out_chunk = P @ V  (M=CH, N=D, K=S), V transposed [b][e][t].
// Wave computes 32x32 (2x2 tiles): 4 WMMAs per 4 fragment loads.
__global__ void pv_kernel(const bf16_t* __restrict__ P, const bf16_t* __restrict__ vbT,
                          float* __restrict__ out, int b, int m_start) {
    int w    = (blockIdx.x * blockDim.x + threadIdx.x) >> 5;
    int lane = threadIdx.x & 31;
    int n0 = (w & 31) << 5;            // 32 N-super-tiles over D
    int m0 = (w >> 5) << 5;            // 16 M-super-tiles over chunk

    const bf16_t* vb = vbT + (size_t)b * D * S;

    v8f acc[2][2] = {};
    for (int kb = 0; kb < S; kb += 32) {
        v16bf a0 = load_a_bf16(P, S, m0,      kb, lane);
        v16bf a1 = load_a_bf16(P, S, m0 + 16, kb, lane);
        v16bf b0 = load_b_bf16(vb, S, n0,      kb, lane);
        v16bf b1 = load_b_bf16(vb, S, n0 + 16, kb, lane);
        acc[0][0] = wmma_bf16(a0, b0, acc[0][0]);
        acc[0][1] = wmma_bf16(a0, b1, acc[0][1]);
        acc[1][0] = wmma_bf16(a1, b0, acc[1][0]);
        acc[1][1] = wmma_bf16(a1, b1, acc[1][1]);
    }

    int roff = (lane >> 4) << 3;
#pragma unroll
    for (int mt = 0; mt < 2; ++mt) {
#pragma unroll
        for (int nt = 0; nt < 2; ++nt) {
            int col = n0 + nt * 16 + (lane & 15);
#pragma unroll
            for (int r = 0; r < 8; ++r) {
                int lr = m0 + mt * 16 + r + roff;
                out[((size_t)b * S + m_start + lr) * D + col] = acc[mt][nt][r];
            }
        }
    }
}

// ---------------- launch ----------------

extern "C" void kernel_launch(void* const* d_in, const int* in_sizes, int n_in,
                              void* d_out, int out_size, void* d_ws, size_t ws_size,
                              hipStream_t stream) {
    const float* x  = (const float*)d_in[0];
    const float* Wq = (const float*)d_in[1];
    const float* bq = (const float*)d_in[2];
    const float* Wk = (const float*)d_in[3];
    const float* bk = (const float*)d_in[4];
    const float* Wv = (const float*)d_in[5];
    const float* bv = (const float*)d_in[6];
    const float* pb = (const float*)d_in[7];
    float* out = (float*)d_out;

    char* ws = (char*)d_ws;
    size_t off = 0;
    auto alloc = [&](size_t bytes) { char* p = ws + off; off += (bytes + 255) & ~(size_t)255; return p; };

    const size_t act = (size_t)NROW * D * sizeof(bf16_t);     // 16 MB each
    bf16_t* cq  = (bf16_t*)alloc(act);
    bf16_t* sq  = (bf16_t*)alloc(act);
    bf16_t* ck  = (bf16_t*)alloc(act);
    bf16_t* sk  = (bf16_t*)alloc(act);
    bf16_t* vbT = (bf16_t*)alloc(act);
    bf16_t* Wqb = (bf16_t*)alloc((size_t)D * D * sizeof(bf16_t));
    bf16_t* Wkb = (bf16_t*)alloc((size_t)D * D * sizeof(bf16_t));
    bf16_t* Wvb = (bf16_t*)alloc((size_t)D * D * sizeof(bf16_t));
    float*  sim = (float*)alloc((size_t)CH * S * sizeof(float));   // 8 MB chunk
    bf16_t* P   = (bf16_t*)alloc((size_t)CH * S * sizeof(bf16_t)); // 4 MB chunk

    // 1) weights to bf16
    int nw = D * D;
    cvt_f32_to_bf16<<<(nw + 255) / 256, 256, 0, stream>>>(Wq, Wqb, nw);
    cvt_f32_to_bf16<<<(nw + 255) / 256, 256, 0, stream>>>(Wk, Wkb, nw);
    cvt_f32_to_bf16<<<(nw + 255) / 256, 256, 0, stream>>>(Wv, Wvb, nw);

    // 2) fused QKV + theta + cos/sin (+ V^T). 512 Mt x 32 Nt32 tiles, 8 waves/block.
    qkv_theta_kernel<<<(512 * 32) / 8, 256, 0, stream>>>(
        x, Wqb, Wkb, Wvb, bq, bk, bv, pb, cq, sq, ck, sk, vbT);

    // 3) per batch / per 512-row query chunk: scores -> softmax -> P@V
    for (int b = 0; b < BATCH; ++b) {
        for (int c = 0; c < S / CH; ++c) {
            int m_start = c * CH;
            scores_kernel<<<(CH / SC_BM) * (S / SC_BN), 256, 0, stream>>>(
                cq, sq, ck, sk, sim, b, m_start);
            softmax_kernel<<<CH, 256, 0, stream>>>(sim, P);
            pv_kernel<<<((CH / 32) * (D / 32)) / 8, 256, 0, stream>>>(
                P, vbT, out, b, m_start);
        }
    }
}
